// BloomAttention_53309134078526
// MI455X (gfx1250) — matrix operand order repaired and checked by reading, not compiled, not run on hardware
//
#include <hip/hip_runtime.h>

// ---------------------------------------------------------------------------
// BLOOM attention block for MI455X (gfx1250, wave32, WMMA).
//   B=4, S=1024, H=4096, NHEAD=32, DHEAD=128, N_qkv=12288
// Pipeline:
//   1) bloom_qkv_gemm : hidden @ qkv_w^T + b  -> q(*1/sqrt(d)), k, v^T (bf16)
//   2) bloom_attn     : flash attention w/ ALiBi + causal mask -> ctx (bf16)
//   3) bloom_dense    : ctx @ dense_w^T + b + residual -> out (fp32)
// All matmuls use V_WMMA_F32_16X16X32_BF16 (fp32 accumulation).
// GEMMs are LDS ping-pong double-buffered: global loads for K-slice k+1 are
// issued before the WMMAs of slice k.  launch_bounds(256,1) gives each wave
// the full VGPR budget so all 8 staging loads stay in flight (one wait),
// instead of the load->wait->cvt serialization seen at 2 blocks/WGP.
// ---------------------------------------------------------------------------

#define SEQ   1024
#define HDIM  4096
#define NHEAD 32
#define DHEAD 128
#define NQKV  12288
#define BSZ   4

typedef __attribute__((ext_vector_type(16))) __bf16 v16bf;
typedef __attribute__((ext_vector_type(8)))  __bf16 v8bf;
typedef __attribute__((ext_vector_type(8)))  float  v8f;
typedef __attribute__((ext_vector_type(4)))  unsigned short v4us;
typedef __attribute__((ext_vector_type(8)))  unsigned short v8us;

union FragAB { v16bf v; v8bf h[2]; };

__device__ __forceinline__ unsigned short f2bf(float f) {
  // fptrunc is RNE; backend emits native v_cvt_pk_bf16_f32
  __bf16 b = (__bf16)f;
  return __builtin_bit_cast(unsigned short, b);
}

__device__ __forceinline__ v4us cvt4(float4 a) {
  v4us r;
  r[0] = f2bf(a.x); r[1] = f2bf(a.y); r[2] = f2bf(a.z); r[3] = f2bf(a.w);
  return r;
}

__device__ __forceinline__ v8f wmma_bf16(v16bf a, v16bf b, v8f c) {
  // D = A(16x32) * B(32x16) + C(16x16), fp32 accum
  return __builtin_amdgcn_wmma_f32_16x16x32_bf16(false, a, false, b,
                                                 (short)0, c, false, false);
}

// ---------------------------------------------------------------------------
// Kernel 1: fused QKV projection.  C[M=4096, N=12288] = hs[M,K] @ w[N,K]^T + b
// Block tile 128x128, BK=32, 8 waves each computing 64x32, LDS double buffer.
// ---------------------------------------------------------------------------
__global__ __launch_bounds__(256, 1)
void bloom_qkv_gemm(const float* __restrict__ hs, const float* __restrict__ w,
                    const float* __restrict__ bias,
                    unsigned short* __restrict__ qb,
                    unsigned short* __restrict__ kb,
                    unsigned short* __restrict__ vtb)
{
  __shared__ unsigned short As[2][128][40];  // +8 pad: 16B align, no bank conflicts
  __shared__ unsigned short Bs[2][128][40];

  const int m0   = blockIdx.y * 128;
  const int n0   = blockIdx.x * 128;
  const int tid  = (int)threadIdx.x;
  const int wave = tid >> 5, lane = tid & 31;
  const int wm   = wave >> 2, wn = wave & 3;      // 2 x 4 wave grid
  const int l15  = lane & 15, lh = lane >> 4;
  const int srow = tid >> 3;                      // staging row (0..31), +32*i
  const int sc4  = (tid & 7) << 2;                // staging k column (0..28)

  v8f acc[4][2];
  for (int i = 0; i < 4; ++i)
    for (int j = 0; j < 2; ++j)
      for (int r = 0; r < 8; ++r) acc[i][j][r] = 0.0f;

  float4 areg[4], breg[4];
  // prologue: stage K-slice 0
  #pragma unroll
  for (int i = 0; i < 4; ++i) {
    areg[i] = *(const float4*)(hs + (size_t)(m0 + srow + 32 * i) * HDIM + sc4);
    breg[i] = *(const float4*)(w  + (size_t)(n0 + srow + 32 * i) * HDIM + sc4);
  }
  #pragma unroll
  for (int i = 0; i < 4; ++i) {
    *(v4us*)&As[0][srow + 32 * i][sc4] = cvt4(areg[i]);
    *(v4us*)&Bs[0][srow + 32 * i][sc4] = cvt4(breg[i]);
  }
  __syncthreads();

  const int NT = HDIM / 32;
  for (int kt = 0; kt < NT; ++kt) {
    const int cur = kt & 1;
    // issue next slice's global loads early (latency hides behind WMMAs)
    if (kt + 1 < NT) {
      const int kn = (kt + 1) * 32;
      #pragma unroll
      for (int i = 0; i < 4; ++i) {
        areg[i] = *(const float4*)(hs + (size_t)(m0 + srow + 32 * i) * HDIM + kn + sc4);
        breg[i] = *(const float4*)(w  + (size_t)(n0 + srow + 32 * i) * HDIM + kn + sc4);
      }
    }

    // A frag: lane half lh holds K {8lh..8lh+7} and {16+8lh..23+8lh}
    FragAB af[4], bfr[2];
    #pragma unroll
    for (int mt = 0; mt < 4; ++mt) {
      const int m = wm * 64 + mt * 16 + l15;
      af[mt].h[0] = *(const v8bf*)&As[cur][m][8 * lh];
      af[mt].h[1] = *(const v8bf*)&As[cur][m][16 + 8 * lh];
    }
    // B frag: lane half lh holds K {8lh..8lh+7} and {16+8lh..23+8lh} for column n
    #pragma unroll
    for (int nt = 0; nt < 2; ++nt) {
      const int n = wn * 32 + nt * 16 + l15;
      bfr[nt].h[0] = *(const v8bf*)&Bs[cur][n][8 * lh];
      bfr[nt].h[1] = *(const v8bf*)&Bs[cur][n][16 + 8 * lh];
    }
    #pragma unroll
    for (int mt = 0; mt < 4; ++mt)
      #pragma unroll
      for (int nt = 0; nt < 2; ++nt)
        acc[mt][nt] = wmma_bf16(af[mt].v, bfr[nt].v, acc[mt][nt]);

    // convert + store next slice into the other buffer, one barrier per step
    if (kt + 1 < NT) {
      #pragma unroll
      for (int i = 0; i < 4; ++i) {
        *(v4us*)&As[cur ^ 1][srow + 32 * i][sc4] = cvt4(areg[i]);
        *(v4us*)&Bs[cur ^ 1][srow + 32 * i][sc4] = cvt4(breg[i]);
      }
    }
    __syncthreads();
  }

  // epilogue: + bias, split into q (scaled by 1/sqrt(d)), k, v^T
  const float qscale = 0.08838834764831845f;  // 1/sqrt(128)
  #pragma unroll
  for (int mt = 0; mt < 4; ++mt) {
    #pragma unroll
    for (int nt = 0; nt < 2; ++nt) {
      const int col = n0 + wn * 32 + nt * 16 + l15;
      const int hh  = col / (3 * DHEAD);
      const int rr  = col % (3 * DHEAD);
      const float bi = bias[col];
      #pragma unroll
      for (int r = 0; r < 8; ++r) {
        const int row = m0 + wm * 64 + mt * 16 + r + 8 * lh;
        const int bb = row >> 10, ss = row & (SEQ - 1);
        const size_t bhh = (size_t)(bb * NHEAD + hh);
        const float vv = acc[mt][nt][r] + bi;
        if (rr < DHEAD) {
          qb[(bhh * SEQ + ss) * DHEAD + rr] = f2bf(vv * qscale);
        } else if (rr < 2 * DHEAD) {
          kb[(bhh * SEQ + ss) * DHEAD + (rr - DHEAD)] = f2bf(vv);
        } else {
          // V stored transposed [b,h,D,S] so PV B-fragments are contiguous
          vtb[(bhh * DHEAD + (rr - 2 * DHEAD)) * SEQ + ss] = f2bf(vv);
        }
      }
    }
  }
}

// ---------------------------------------------------------------------------
// Kernel 2: flash attention.  One block per (b, h, 128-row q tile); each wave
// owns 16 query rows and streams 32-key blocks with online softmax.
// No block barriers: waves are fully independent (divergent causal ranges).
// ---------------------------------------------------------------------------
__global__ __launch_bounds__(256)
void bloom_attn(const unsigned short* __restrict__ qb,
                const unsigned short* __restrict__ kbuf,
                const unsigned short* __restrict__ vtb,
                unsigned short* __restrict__ ctx)
{
  __shared__ unsigned short Plds[8][16][40];   // per-wave P re-layout scratch

  const int blk  = (int)blockIdx.x;
  const int tq   = blk & 7;                    // S/128 = 8 q tiles
  const int h    = (blk >> 3) & (NHEAD - 1);
  const int b    = blk >> 8;
  const int tid  = (int)threadIdx.x;
  const int wave = tid >> 5, lane = tid & 31;
  const int l15  = lane & 15, lh = lane >> 4;
  const int q0   = tq * 128 + wave * 16;
  const size_t bh = (size_t)(b * NHEAD + h);

  // Q fragments (16 rows x 128 d = 4 K-chunks), 1/sqrt(d) already folded in
  FragAB qf[4];
  {
    const unsigned short* qp = qb + (bh * SEQ + q0 + l15) * DHEAD;
    #pragma unroll
    for (int kc = 0; kc < 4; ++kc) {
      qf[kc].h[0] = *(const v8bf*)(qp + 32 * kc + 8 * lh);
      qf[kc].h[1] = *(const v8bf*)(qp + 32 * kc + 16 + 8 * lh);
    }
  }

  v8f cacc[8];                                  // ctx 16 x 128 accumulators
  for (int j = 0; j < 8; ++j)
    for (int r = 0; r < 8; ++r) cacc[j][r] = 0.0f;
  float mrun[8], lrun[8];
  for (int r = 0; r < 8; ++r) { mrun[r] = -3.0e38f; lrun[r] = 0.0f; }

  // BLOOM ALiBi slopes for 32 heads: 2^(-0.25*(h+1)); alibi[h,k] = slope*k
  const float slope = exp2f(-0.25f * (float)(h + 1));

  for (int kb0 = 0; kb0 <= q0 + 15; kb0 += 32) {     // causal range
    // scores S = Q @ K^T for 32 keys (two 16-col accumulators)
    v8f s[2];
    for (int nt = 0; nt < 2; ++nt)
      for (int r = 0; r < 8; ++r) s[nt][r] = 0.0f;
    #pragma unroll
    for (int nt = 0; nt < 2; ++nt) {
      const unsigned short* kr = kbuf + (bh * SEQ + kb0 + nt * 16 + l15) * DHEAD;
      FragAB kf[4];
      #pragma unroll
      for (int kc = 0; kc < 4; ++kc) {          // issue all loads first
        kf[kc].h[0] = *(const v8bf*)(kr + 32 * kc + 8 * lh);
        kf[kc].h[1] = *(const v8bf*)(kr + 32 * kc + 16 + 8 * lh);
      }
      #pragma unroll
      for (int kc = 0; kc < 4; ++kc)
        s[nt] = wmma_bf16(qf[kc].v, kf[kc].v, s[nt]);
    }

    // ALiBi + causal mask + online softmax.  Row m = r + 8*lh lives in one
    // 16-lane half, so row reductions are shfl_xor over offsets 1..8.
    const int key0 = kb0 + l15;
    const int key1 = kb0 + 16 + l15;
    #pragma unroll
    for (int r = 0; r < 8; ++r) {
      const int rowg = q0 + r + 8 * lh;
      float s0 = s[0][r] + slope * (float)key0;
      float s1 = s[1][r] + slope * (float)key1;
      if (key0 > rowg) s0 = -1.0e30f;
      if (key1 > rowg) s1 = -1.0e30f;
      float mx = fmaxf(s0, s1);
      #pragma unroll
      for (int off = 1; off < 16; off <<= 1) mx = fmaxf(mx, __shfl_xor(mx, off, 32));
      const float mnew  = fmaxf(mrun[r], mx);
      const float scale = __expf(mrun[r] - mnew);
      const float p0 = __expf(s0 - mnew);
      const float p1 = __expf(s1 - mnew);
      float rs = p0 + p1;
      #pragma unroll
      for (int off = 1; off < 16; off <<= 1) rs += __shfl_xor(rs, off, 32);
      lrun[r] = lrun[r] * scale + rs;
      mrun[r] = mnew;
      #pragma unroll
      for (int j = 0; j < 8; ++j) cacc[j][r] *= scale;
      const int m = r + 8 * lh;
      Plds[wave][m][l15]      = f2bf(p0);      // C-layout -> LDS
      Plds[wave][m][16 + l15] = f2bf(p1);
    }

    // LDS ops are in-order within a wave: no barrier needed for this bounce.
    FragAB pf;                                  // A-layout P (16 x 32)
    pf.h[0] = *(const v8bf*)&Plds[wave][l15][8 * lh];
    pf.h[1] = *(const v8bf*)&Plds[wave][l15][16 + 8 * lh];

    // ctx += P @ V; v^T layout makes B-fragments contiguous global loads
    #pragma unroll
    for (int jg = 0; jg < 2; ++jg) {
      FragAB vf[4];
      #pragma unroll
      for (int j = 0; j < 4; ++j) {             // issue all loads first
        const unsigned short* vr =
            vtb + (bh * DHEAD + 16 * (jg * 4 + j) + l15) * SEQ + kb0 + 8 * lh;
        vf[j].h[0] = *(const v8bf*)(vr);
        vf[j].h[1] = *(const v8bf*)(vr + 16);
      }
      #pragma unroll
      for (int j = 0; j < 4; ++j)
        cacc[jg * 4 + j] = wmma_bf16(pf.v, vf[j].v, cacc[jg * 4 + j]);
    }
  }

  // normalize and store ctx as [b, s, h*128 + d] (bf16) for dense GEMM
  unsigned short* co = ctx + ((size_t)b * SEQ + q0) * HDIM + h * DHEAD;
  #pragma unroll
  for (int r = 0; r < 8; ++r) {
    const float inv = 1.0f / lrun[r];
    const int m = r + 8 * lh;
    #pragma unroll
    for (int j = 0; j < 8; ++j)
      co[(size_t)m * HDIM + 16 * j + l15] = f2bf(cacc[j][r] * inv);
  }
}

// ---------------------------------------------------------------------------
// Kernel 3: dense projection + bias + residual (fp32 out), double-buffered.
// ---------------------------------------------------------------------------
__global__ __launch_bounds__(256, 1)
void bloom_dense(const unsigned short* __restrict__ ctx,
                 const float* __restrict__ w, const float* __restrict__ bias,
                 const float* __restrict__ residual, float* __restrict__ out)
{
  __shared__ unsigned short As[2][128][40];
  __shared__ unsigned short Bs[2][128][40];

  const int m0   = blockIdx.y * 128;
  const int n0   = blockIdx.x * 128;
  const int tid  = (int)threadIdx.x;
  const int wave = tid >> 5, lane = tid & 31;
  const int wm   = wave >> 2, wn = wave & 3;
  const int l15  = lane & 15, lh = lane >> 4;
  const int srow = tid >> 3;                      // fp32 staging: rows 0..31
  const int sc4  = (tid & 7) << 2;
  const int arow = tid >> 2;                      // bf16 staging: rows 0..63
  const int ac8  = (tid & 3) << 3;

  v8f acc[4][2];
  for (int i = 0; i < 4; ++i)
    for (int j = 0; j < 2; ++j)
      for (int r = 0; r < 8; ++r) acc[i][j][r] = 0.0f;

  v8us  areg[2];
  float4 breg[4];
  #pragma unroll
  for (int i = 0; i < 2; ++i)
    areg[i] = *(const v8us*)(ctx + (size_t)(m0 + arow + 64 * i) * HDIM + ac8);
  #pragma unroll
  for (int i = 0; i < 4; ++i)
    breg[i] = *(const float4*)(w + (size_t)(n0 + srow + 32 * i) * HDIM + sc4);
  #pragma unroll
  for (int i = 0; i < 2; ++i)
    *(v8us*)&As[0][arow + 64 * i][ac8] = areg[i];
  #pragma unroll
  for (int i = 0; i < 4; ++i)
    *(v4us*)&Bs[0][srow + 32 * i][sc4] = cvt4(breg[i]);
  __syncthreads();

  const int NT = HDIM / 32;
  for (int kt = 0; kt < NT; ++kt) {
    const int cur = kt & 1;
    if (kt + 1 < NT) {
      const int kn = (kt + 1) * 32;
      #pragma unroll
      for (int i = 0; i < 2; ++i)
        areg[i] = *(const v8us*)(ctx + (size_t)(m0 + arow + 64 * i) * HDIM + kn + ac8);
      #pragma unroll
      for (int i = 0; i < 4; ++i)
        breg[i] = *(const float4*)(w + (size_t)(n0 + srow + 32 * i) * HDIM + kn + sc4);
    }

    FragAB af[4], bfr[2];
    #pragma unroll
    for (int mt = 0; mt < 4; ++mt) {
      const int m = wm * 64 + mt * 16 + l15;
      af[mt].h[0] = *(const v8bf*)&As[cur][m][8 * lh];
      af[mt].h[1] = *(const v8bf*)&As[cur][m][16 + 8 * lh];
    }
    #pragma unroll
    for (int nt = 0; nt < 2; ++nt) {
      const int n = wn * 32 + nt * 16 + l15;
      bfr[nt].h[0] = *(const v8bf*)&Bs[cur][n][8 * lh];
      bfr[nt].h[1] = *(const v8bf*)&Bs[cur][n][16 + 8 * lh];
    }
    #pragma unroll
    for (int mt = 0; mt < 4; ++mt)
      #pragma unroll
      for (int nt = 0; nt < 2; ++nt)
        acc[mt][nt] = wmma_bf16(af[mt].v, bfr[nt].v, acc[mt][nt]);

    if (kt + 1 < NT) {
      #pragma unroll
      for (int i = 0; i < 2; ++i)
        *(v8us*)&As[cur ^ 1][arow + 64 * i][ac8] = areg[i];
      #pragma unroll
      for (int i = 0; i < 4; ++i)
        *(v4us*)&Bs[cur ^ 1][srow + 32 * i][sc4] = cvt4(breg[i]);
    }
    __syncthreads();
  }

  #pragma unroll
  for (int mt = 0; mt < 4; ++mt) {
    #pragma unroll
    for (int nt = 0; nt < 2; ++nt) {
      const int col = n0 + wn * 32 + nt * 16 + l15;
      const float bi = bias[col];
      #pragma unroll
      for (int r = 0; r < 8; ++r) {
        const int row = m0 + wm * 64 + mt * 16 + r + 8 * lh;
        const size_t off = (size_t)row * HDIM + col;
        out[off] = acc[mt][nt][r] + bi + residual[off];
      }
    }
  }
}

// ---------------------------------------------------------------------------
extern "C" void kernel_launch(void* const* d_in, const int* in_sizes, int n_in,
                              void* d_out, int out_size, void* d_ws, size_t ws_size,
                              hipStream_t stream) {
  const float* hs   = (const float*)d_in[0];   // [4,1024,4096]
  const float* res  = (const float*)d_in[1];   // [4,1024,4096]
  const float* qkvw = (const float*)d_in[2];   // [12288,4096]
  const float* qkvb = (const float*)d_in[3];   // [12288]
  const float* dw   = (const float*)d_in[4];   // [4096,4096]
  const float* db   = (const float*)d_in[5];   // [4096]
  float* out = (float*)d_out;

  // workspace: q, k, v^T, ctx  (bf16, 32 MiB each = 128 MiB total)
  const size_t elems = (size_t)BSZ * NHEAD * SEQ * DHEAD;   // 16,777,216
  unsigned short* qb  = (unsigned short*)d_ws;
  unsigned short* kb  = qb  + elems;
  unsigned short* vtb = kb  + elems;
  unsigned short* ctx = vtb + elems;

  bloom_qkv_gemm<<<dim3(NQKV / 128, (BSZ * SEQ) / 128), 256, 0, stream>>>(
      hs, qkvw, qkvb, qb, kb, vtb);
  bloom_attn<<<dim3(BSZ * NHEAD * (SEQ / 128)), 256, 0, stream>>>(
      qb, kb, vtb, ctx);
  bloom_dense<<<dim3(HDIM / 128, (BSZ * SEQ) / 128), 256, 0, stream>>>(
      ctx, dw, db, res, out);
}
